// NonLocalBlock_62526133895211
// MI455X (gfx1250) — matrix operand order repaired and checked by reading, not compiled, not run on hardware
//
#include <hip/hip_runtime.h>
#include <hip/hip_bf16.h>

// NonLocalBlock for MI455X (gfx1250): GroupNorm -> QKV -> N^2 attention -> proj+residual
// All GEMMs use v_wmma_f32_16x16x32_bf16 (bf16 in, f32 accumulate) with 2x4 register
// blocking per wave (8 independent accumulators -> ~21 FLOP/B from L2, hazard-free
// WMMA bursts). Per-batch scores (67MB f32) + weights (33.5MB bf16) stay L2-resident.

typedef __bf16 bf16;
typedef __attribute__((ext_vector_type(16))) bf16 v16bf;
typedef __attribute__((ext_vector_type(8)))  bf16 bf16x8;
typedef __attribute__((ext_vector_type(8)))  float v8f;

#define Cc 256
#define NG 32
#define NSP 4096   // H*W
#define NB 4

// ---------- fragment loaders (wave32 WMMA VGPR layouts, CDNA5 ISA) ----------
// A: 16x32 bf16, row-major source, rows m0..m0+15, k = kb..kb+31.
// lane L (half=L>>4, r=L&15): row = m0+r; elements 0..7 = k(kb+8*half ..+7), 8..15 = +16.
static __device__ __forceinline__ v16bf load_frag_a(const bf16* __restrict__ A, int lda,
                                                    int m0, int kb, int lane) {
  const int half = lane >> 4, r = lane & 15;
  const bf16* p = A + (size_t)(m0 + r) * lda + kb + half * 8;
  bf16x8 lo = *(const bf16x8*)p;
  bf16x8 hi = *(const bf16x8*)(p + 16);
  v16bf f;
#pragma unroll
  for (int i = 0; i < 8; ++i) { f[i] = lo[i]; f[i + 8] = hi[i]; }
  return f;
}

// B: 32x16 bf16 taken from Bt (the transposed matrix, row-major N x K):
// lane holds column n = n0 + (L&15), elements 0..7 = k(kb+8*(L>>4) ..+7), 8..15 = +16.
static __device__ __forceinline__ v16bf load_frag_b(const bf16* __restrict__ Bt, int ldb,
                                                    int n0, int kb, int lane) {
  const int col = n0 + (lane & 15);
  const int koff = kb + (lane >> 4) * 8;
  const bf16* p = Bt + (size_t)col * ldb + koff;
  bf16x8 lo = *(const bf16x8*)p;
  bf16x8 hi = *(const bf16x8*)(p + 16);
  v16bf f;
#pragma unroll
  for (int i = 0; i < 8; ++i) { f[i] = lo[i]; f[i + 8] = hi[i]; }
  return f;
}

// ---------- kernels ----------
__global__ void f32_to_bf16_kernel(const float* __restrict__ src, bf16* __restrict__ dst, int n) {
  int i = blockIdx.x * blockDim.x + threadIdx.x;
  if (i < n) dst[i] = (bf16)src[i];
}

// One block per (batch, group). Emits xn (f32, C x N per batch) and xh_t (bf16, N x C per batch).
__global__ void groupnorm_kernel(const float* __restrict__ x, const float* __restrict__ gw,
                                 const float* __restrict__ gb, float* __restrict__ xn,
                                 bf16* __restrict__ xht) {
  const int b = blockIdx.x / NG, g = blockIdx.x % NG;
  const int cpg = Cc / NG;                 // 8
  const int M = cpg * NSP;                 // 32768
  const size_t base = ((size_t)b * Cc + (size_t)g * cpg) * NSP;
  const int tid = threadIdx.x;

  float s = 0.f, ss = 0.f;
  for (int i = tid; i < M; i += blockDim.x) {
    float v = x[base + i];
    s += v; ss += v * v;
  }
  __shared__ float r0[256], r1[256];
  r0[tid] = s; r1[tid] = ss;
  __syncthreads();
  for (int st = 128; st > 0; st >>= 1) {
    if (tid < st) { r0[tid] += r0[tid + st]; r1[tid] += r1[tid + st]; }
    __syncthreads();
  }
  const float mean = r0[0] / (float)M;
  const float var  = r1[0] / (float)M - mean * mean;
  const float inv  = rsqrtf(var + 1e-5f);

  for (int i = tid; i < M; i += blockDim.x) {
    int cl = i >> 12;            // i / NSP
    int n  = i & (NSP - 1);      // i % NSP
    int c  = g * cpg + cl;
    float v = (x[base + i] - mean) * inv * gw[c] + gb[c];
    xn[base + i] = v;
    xht[((size_t)b * NSP + n) * Cc + c] = (bf16)v;
  }
}

// Generic bf16 WMMA GEMM with MT x NT register blocking: D(M x Nn) = A(M x K) * B(K x Nn),
// B given transposed (Bt: Nn x K). 8 wave32 per block; each wave owns an MT*16 x NT*16 block.
// mode 0: bf16 store transposed (out[n*ldo+m]) + optional bias[m]
// mode 1: bf16 store direct    (out[m*ldo+n]) + optional bias[m]
// mode 2: f32 store direct, scaled
// mode 3: f32 store direct + bias[m] + resid[m*ldo+n]
template <int MT, int NT>
__global__ void gemm_wmma_bf16(const bf16* __restrict__ A, int lda,
                               const bf16* __restrict__ Bt, int ldb,
                               int M, int Nn, int K,
                               const float* __restrict__ bias,
                               const float* __restrict__ resid, float scale,
                               bf16* __restrict__ outB, float* __restrict__ outF,
                               int ldo, int mode) {
  const int lane = threadIdx.x & 31;
  const int wave = threadIdx.x >> 5;
  const long job = (long)blockIdx.x * (blockDim.x >> 5) + wave;
  const int mBlocks = (M >> 4) / MT;
  const int nBlocks = (Nn >> 4) / NT;
  if (job >= (long)mBlocks * nBlocks) return;  // uniform per wave: EXEC all-1s for WMMA
  const int m0 = (int)(job % mBlocks) * (16 * MT);
  const int n0 = (int)(job / mBlocks) * (16 * NT);

  v8f acc[MT][NT];
#pragma unroll
  for (int mi = 0; mi < MT; ++mi)
#pragma unroll
    for (int ni = 0; ni < NT; ++ni) acc[mi][ni] = (v8f){};

  for (int k = 0; k < K; k += 32) {
    if (k + 32 < K) {  // gfx1250 global_prefetch_b8: lanes spread over the block's rows/cols
      __builtin_prefetch(A  + (size_t)(m0 + (lane & (16 * MT - 1))) * lda + k + 32, 0, 1);
      __builtin_prefetch(Bt + (size_t)(n0 + lane) * ldb + k + 32, 0, 1);
      if (NT > 2)
        __builtin_prefetch(Bt + (size_t)(n0 + 32 + lane) * ldb + k + 32, 0, 1);
    }
    v16bf a[MT], b[NT];
#pragma unroll
    for (int mi = 0; mi < MT; ++mi) a[mi] = load_frag_a(A, lda, m0 + 16 * mi, k, lane);
#pragma unroll
    for (int ni = 0; ni < NT; ++ni) b[ni] = load_frag_b(Bt, ldb, n0 + 16 * ni, k, lane);
#pragma unroll
    for (int mi = 0; mi < MT; ++mi)
#pragma unroll
      for (int ni = 0; ni < NT; ++ni)
        acc[mi][ni] = __builtin_amdgcn_wmma_f32_16x16x32_bf16(
            false, a[mi], false, b[ni], (short)0, acc[mi][ni], false, false);
  }

  const int half = lane >> 4;
  const int col = lane & 15;
#pragma unroll
  for (int mi = 0; mi < MT; ++mi) {
#pragma unroll
    for (int ni = 0; ni < NT; ++ni) {
      const int n = n0 + 16 * ni + col;
      const int mb = m0 + 16 * mi + 8 * half;
      v8f aa = acc[mi][ni];
      if (mode == 0) {             // transposed bf16 store: one b128 per lane
        bf16x8 pk;
#pragma unroll
        for (int e = 0; e < 8; ++e) {
          float v = aa[e] + (bias ? bias[mb + e] : 0.f);
          pk[e] = (bf16)v;
        }
        *(bf16x8*)(outB + (size_t)n * ldo + mb) = pk;
      } else if (mode == 1) {      // direct bf16 store
#pragma unroll
        for (int e = 0; e < 8; ++e)
          outB[(size_t)(mb + e) * ldo + n] = (bf16)(aa[e] + (bias ? bias[mb + e] : 0.f));
      } else if (mode == 2) {      // scaled f32 (attention scores)
#pragma unroll
        for (int e = 0; e < 8; ++e)
          outF[(size_t)(mb + e) * ldo + n] = aa[e] * scale;
      } else {                     // f32 + bias + residual (final output)
#pragma unroll
        for (int e = 0; e < 8; ++e) {
          size_t idx = (size_t)(mb + e) * ldo + n;
          outF[idx] = resid[idx] + aa[e] + bias[mb + e];
        }
      }
    }
  }
}

// Row softmax: one block per row of scores (4096 f32), output bf16 row (B-operand layout for att GEMM).
__global__ void softmax_rows_kernel(const float* __restrict__ S, bf16* __restrict__ W) {
  const int row = blockIdx.x;
  const int tid = threadIdx.x;
  const float* s = S + (size_t)row * NSP;
  __shared__ float buf[NSP];
  __shared__ float red[256];

  float mx = -3.4e38f;
  for (int i = tid; i < NSP; i += 256) {
    float v = s[i];
    buf[i] = v;
    mx = fmaxf(mx, v);
  }
  red[tid] = mx; __syncthreads();
  for (int st = 128; st > 0; st >>= 1) {
    if (tid < st) red[tid] = fmaxf(red[tid], red[tid + st]);
    __syncthreads();
  }
  mx = red[0];
  __syncthreads();

  float sum = 0.f;
  for (int i = tid; i < NSP; i += 256) {
    float e = __expf(buf[i] - mx);
    buf[i] = e;
    sum += e;
  }
  red[tid] = sum; __syncthreads();
  for (int st = 128; st > 0; st >>= 1) {
    if (tid < st) red[tid] += red[tid + st];
    __syncthreads();
  }
  const float rinv = 1.f / red[0];
  for (int i = tid; i < NSP; i += 256)
    W[(size_t)row * NSP + i] = (bf16)(buf[i] * rinv);
}

// ---------- host ----------
extern "C" void kernel_launch(void* const* d_in, const int* in_sizes, int n_in,
                              void* d_out, int out_size, void* d_ws, size_t ws_size,
                              hipStream_t stream) {
  const float* x    = (const float*)d_in[0];
  const float* gn_w = (const float*)d_in[1];
  const float* gn_b = (const float*)d_in[2];
  const float* qw   = (const float*)d_in[3];
  const float* qb   = (const float*)d_in[4];
  const float* kw   = (const float*)d_in[5];
  const float* kb   = (const float*)d_in[6];
  const float* vw   = (const float*)d_in[7];
  const float* vb   = (const float*)d_in[8];
  const float* ow   = (const float*)d_in[9];
  const float* ob   = (const float*)d_in[10];
  float* out = (float*)d_out;

  // workspace carve-up (~129 MB total)
  char* p = (char*)d_ws;
  size_t off = 0;
  auto carve = [&](size_t bytes) -> char* {
    char* r = p + off;
    off = (off + bytes + 255) & ~(size_t)255;
    return r;
  };
  bf16*  wq_h  = (bf16*)carve((size_t)Cc * Cc * 2);
  bf16*  wk_h  = (bf16*)carve((size_t)Cc * Cc * 2);
  bf16*  wv_h  = (bf16*)carve((size_t)Cc * Cc * 2);
  bf16*  wo_h  = (bf16*)carve((size_t)Cc * Cc * 2);
  float* xn    = (float*)carve((size_t)NB * Cc * NSP * 4);
  bf16*  xht   = (bf16*) carve((size_t)NB * NSP * Cc * 2);
  bf16*  qt    = (bf16*) carve((size_t)NSP * Cc * 2);
  bf16*  kt    = (bf16*) carve((size_t)NSP * Cc * 2);
  bf16*  vmat  = (bf16*) carve((size_t)Cc * NSP * 2);
  float* scores= (float*)carve((size_t)NSP * NSP * 4);
  bf16*  wts   = (bf16*) carve((size_t)NSP * NSP * 2);
  bf16*  att_t = (bf16*) carve((size_t)NSP * Cc * 2);
  (void)ws_size; (void)in_sizes; (void)n_in; (void)out_size;

  // weights -> bf16
  const int WN = Cc * Cc;
  f32_to_bf16_kernel<<<WN / 256, 256, 0, stream>>>(qw, wq_h, WN);
  f32_to_bf16_kernel<<<WN / 256, 256, 0, stream>>>(kw, wk_h, WN);
  f32_to_bf16_kernel<<<WN / 256, 256, 0, stream>>>(vw, wv_h, WN);
  f32_to_bf16_kernel<<<WN / 256, 256, 0, stream>>>(ow, wo_h, WN);

  // GroupNorm: xn (f32 resid base) + xh_t (bf16, N x C per batch)
  groupnorm_kernel<<<NB * NG, 256, 0, stream>>>(x, gn_w, gn_b, xn, xht);

  // wave-jobs (MT=2, NT=4): QKV/att/out: (16/2)*(256/4)=512 -> 64 blocks of 8 waves
  //                         scores: (256/2)*(256/4)=8192 -> 1024 blocks
  const int blocksQKV = ((Cc / 16 / 2) * (NSP / 16 / 4)) / 8;
  const int blocksSc  = ((NSP / 16 / 2) * (NSP / 16 / 4)) / 8;

  for (int b = 0; b < NB; ++b) {
    const bf16* xhtb = xht + (size_t)b * NSP * Cc;
    float* xnb  = xn  + (size_t)b * Cc * NSP;
    float* outb = out + (size_t)b * Cc * NSP;

    // Q^T, K^T (N x C bf16), V (C x N bf16)
    gemm_wmma_bf16<2, 4><<<blocksQKV, 256, 0, stream>>>(wq_h, Cc, xhtb, Cc, Cc, NSP, Cc,
                                                        qb, nullptr, 1.f, qt, nullptr, Cc, 0);
    gemm_wmma_bf16<2, 4><<<blocksQKV, 256, 0, stream>>>(wk_h, Cc, xhtb, Cc, Cc, NSP, Cc,
                                                        kb, nullptr, 1.f, kt, nullptr, Cc, 0);
    gemm_wmma_bf16<2, 4><<<blocksQKV, 256, 0, stream>>>(wv_h, Cc, xhtb, Cc, Cc, NSP, Cc,
                                                        vb, nullptr, 1.f, vmat, nullptr, NSP, 1);

    // scores[i,j] = (q^T k)/sqrt(C): A = qt (N x C), Bt = kt (N x C) -> f32 N x N
    gemm_wmma_bf16<2, 4><<<blocksSc, 256, 0, stream>>>(qt, Cc, kt, Cc, NSP, NSP, Cc,
                                                       nullptr, nullptr, 0.0625f,
                                                       nullptr, scores, NSP, 2);

    // softmax over keys -> bf16 weights (row-major i x j, exactly the Bt layout for att)
    softmax_rows_kernel<<<NSP, 256, 0, stream>>>(scores, wts);

    // att[c,i] = V(c,j) W^T(j,i): A = vmat (C x N), Bt = wts (N_i x N_j); store att^T (N x C)
    gemm_wmma_bf16<2, 4><<<blocksQKV, 256, 0, stream>>>(vmat, NSP, wts, NSP, Cc, NSP, NSP,
                                                        nullptr, nullptr, 1.f, att_t, nullptr, Cc, 0);

    // out = xn + ow @ att + ob
    gemm_wmma_bf16<2, 4><<<blocksQKV, 256, 0, stream>>>(wo_h, Cc, att_t, Cc, Cc, NSP, Cc,
                                                        ob, xnb, 1.f, nullptr, outb, NSP, 3);
  }
}
